// SelPredictor_26817775797025
// MI455X (gfx1250) — matrix operand order repaired and checked by reading, not compiled, not run on hardware
//
#include <hip/hip_runtime.h>
#include <stdint.h>
#include <stddef.h>

// ===========================================================================
// SelPredictor forward on MI455X (gfx1250, wave32, WMMA).
// Heavy matmuls -> v_wmma_f32_16x16x32_f16. One wave computes a 16x(16*NT)
// strip so a single A fragment feeds NT WMMAs (A-operand register reuse);
// per-lane fragments use contiguous k = 16*(lane>>4)+e for A and B so all operand loads
// are K-contiguous b128s.
// ===========================================================================

typedef _Float16 half_t;
typedef __attribute__((ext_vector_type(16))) _Float16 v16h;
typedef __attribute__((ext_vector_type(8)))  _Float16 v8h;
typedef __attribute__((ext_vector_type(8)))  float    v8f;

#define BB    128   // batch
#define TT    128   // question length
#define LN_   12    // column-name length
#define NW_   12    // word embedding dim
#define NH_   512   // bi-LSTM output (2*H)
#define HH_   256   // per-direction hidden
#define G4_   1024  // 4*H gates
#define CM_   48    // max columns per sample
#define NS_   5     // selected columns
#define KP_   16    // selected columns padded to one 16-row WMMA tile

// ------------------------------ utilities ---------------------------------
__global__ void k_zero(uint32_t* __restrict__ p, long n) {
  long i = (long)blockIdx.x * blockDim.x + threadIdx.x;
  if (i < n) p[i] = 0u;
}

__global__ void k_cvt(const float* __restrict__ s, half_t* __restrict__ d, long n) {
  long i = (long)blockIdx.x * blockDim.x + threadIdx.x;
  if (i < n) d[i] = (half_t)s[i];
}

// Pack [Wih | Whh] (f32) into one f16 [1024, din_p+256] matrix, zero-padding
// the input slice from din to din_p (multiple of 32 for the WMMA K loop).
__global__ void k_pack_wcat(const float* __restrict__ Wih, const float* __restrict__ Whh,
                            half_t* __restrict__ dst, int din, int din_p) {
  const int dK = din_p + HH_;
  long i = (long)blockIdx.x * blockDim.x + threadIdx.x;
  if (i >= (long)G4_ * dK) return;
  const int o = (int)(i / dK), c = (int)(i % dK);
  float v = 0.f;
  if (c < din_p) { if (c < din) v = Wih[(long)o * din + c]; }
  else            v = Whh[(long)o * HH_ + (c - din_p)];
  dst[i] = (half_t)v;
}

// ------------------------------ WMMA GEMMs --------------------------------
// C[M,N](f32, opt f16 copy) = A[M,K]f16 * W[N,K]^T f16 (+bias[N]); batched by z.
// One wave = 16x(16*NT) strip (A fragment reused across NT WMMAs); block = 4
// waves across N. M%16==0, K%32==0, N%(64*NT)==0 for the chosen NT.
// 16-bit fragment k order used here (contiguous, same for A and B):
//   A: m=lane&15; e    -> k=(lane>>4)*16+e
//   B: n=lane&15; e    -> k=(lane>>4)*16+e
// -> per-lane loads are K-contiguous: A = 2x b128, B = 1x b256 per tile.
template <int NT>
__global__ void k_gemm_tn(const half_t* __restrict__ A, const half_t* __restrict__ W,
                          const float* __restrict__ bias,
                          float* __restrict__ C, half_t* __restrict__ C16,
                          int M, int N, int K, long sA, long sW, long sC) {
  A += (long)blockIdx.z * sA;
  W += (long)blockIdx.z * sW;
  const long cb = (long)blockIdx.z * sC;
  const int wave = threadIdx.x >> 5, lane = threadIdx.x & 31;
  const int m0 = blockIdx.x << 4;
  const int nbase = (blockIdx.y * 4 + wave) * (16 * NT);
  if (nbase >= N) return;
  const int lo = lane & 15, hi = lane >> 4;
  const half_t* pa = A + (long)(m0 + lo) * K + (hi << 4);
  const half_t* pb[NT];
  #pragma unroll
  for (int nt = 0; nt < NT; ++nt)
    pb[nt] = W + (long)(nbase + nt * 16 + lo) * K + (hi << 4);
  v8f acc[NT];
  #pragma unroll
  for (int nt = 0; nt < NT; ++nt) acc[nt] = (v8f){};
  for (int k = 0; k < K; k += 32) {
    __builtin_prefetch(pa + k + 128, 0, 1);        // global_prefetch_b8
    __builtin_prefetch(pb[0] + k + 128, 0, 1);
    v8h a0 = *(const v8h*)(pa + k);
    v8h a1 = *(const v8h*)(pa + k + 8);
    v16h a = __builtin_shufflevector(a0, a1, 0,1,2,3,4,5,6,7,8,9,10,11,12,13,14,15);
    #pragma unroll
    for (int nt = 0; nt < NT; ++nt) {
      v16h b = *(const v16h*)(pb[nt] + k);
      acc[nt] = __builtin_amdgcn_wmma_f32_16x16x32_f16(false, a, false, b, (short)0,
                                                       acc[nt], false, false);
    }
  }
  const int mr = m0 + (hi << 3);
  #pragma unroll
  for (int nt = 0; nt < NT; ++nt) {
    const int n = nbase + nt * 16 + lo;
    const float bv = bias ? bias[n] : 0.f;
    #pragma unroll
    for (int r = 0; r < 8; ++r) {
      long o = cb + (long)(mr + r) * N + n;
      float v = acc[nt][r] + bv;
      C[o] = v;
      if (C16) C16[o] = (half_t)v;
    }
  }
}

// C[M,N] = A[M,K]f32 * B[K,N]f16 (both row-major); batched by z. Used for
// prob x q_enc context sums (A = softmax output in f32, converted in-register;
// the conversion cost is amortized over NT WMMAs).
template <int NT>
__global__ void k_gemm_nn(const float* __restrict__ A, const half_t* __restrict__ Bm,
                          float* __restrict__ C, half_t* __restrict__ C16,
                          int M, int N, int K, long sA, long sB, long sC) {
  A  += (long)blockIdx.z * sA;
  Bm += (long)blockIdx.z * sB;
  const long cb = (long)blockIdx.z * sC;
  const int wave = threadIdx.x >> 5, lane = threadIdx.x & 31;
  const int m0 = blockIdx.x << 4;
  const int nbase = (blockIdx.y * 4 + wave) * (16 * NT);
  if (nbase >= N) return;
  const int lo = lane & 15, hi = lane >> 4;
  const float* pa = A + (long)(m0 + lo) * K + (hi << 4);
  v8f acc[NT];
  #pragma unroll
  for (int nt = 0; nt < NT; ++nt) acc[nt] = (v8f){};
  for (int k = 0; k < K; k += 32) {
    v16h a;
    #pragma unroll
    for (int e = 0; e < 8; ++e) {
      a[e]     = (half_t)pa[k + e];
      a[e + 8] = (half_t)pa[k + 8 + e];
    }
    const half_t* pbk = Bm + (long)(k + (hi << 4)) * N + nbase + lo;
    #pragma unroll
    for (int nt = 0; nt < NT; ++nt) {
      v16h b;
      #pragma unroll
      for (int e = 0; e < 16; ++e) b[e] = pbk[(long)e * N + nt * 16];
      acc[nt] = __builtin_amdgcn_wmma_f32_16x16x32_f16(false, a, false, b, (short)0,
                                                       acc[nt], false, false);
    }
  }
  const int mr = m0 + (hi << 3);
  #pragma unroll
  for (int nt = 0; nt < NT; ++nt) {
    #pragma unroll
    for (int r = 0; r < 8; ++r) {
      long o = cb + (long)(mr + r) * N + nbase + nt * 16 + lo;
      C[o] = acc[nt][r];
      if (C16) C16[o] = (half_t)acc[nt][r];
    }
  }
}

// ------------------------------ LSTM pieces -------------------------------
// abuf[r, 0:din_p) = x_t (f32 emb or f16 prev-layer out, time-reversed for bw
// with packed-sequence semantics), abuf[r, din_p:din_p+256) = h_{t-1}.
__global__ void k_pack_step(const float* __restrict__ s32, const half_t* __restrict__ s16,
                            const half_t* __restrict__ h16, half_t* __restrict__ abuf,
                            const int* __restrict__ lens,
                            int nSeq, int Rpad, int T, int dinSrc, int din_p,
                            int t, int dir) {
  const int dK = din_p + HH_;
  long i = (long)blockIdx.x * blockDim.x + threadIdx.x;
  if (i >= (long)Rpad * dK) return;
  const int r = (int)(i / dK), c = (int)(i % dK);
  half_t v = (half_t)0.f;
  if (r < nSeq) {
    if (c < din_p) {
      if (c < dinSrc) {
        const int len = lens[r];
        const int ts = dir ? (len - 1 - t) : t;
        const bool valid = dir ? (t < len) : true;
        if (valid) {
          long o = ((long)r * T + ts) * dinSrc + c;
          v = s32 ? (half_t)s32[o] : s16[o];
        }
      }
    } else {
      v = h16[(long)r * HH_ + (c - din_p)];
    }
  }
  abuf[i] = v;
}

__device__ __forceinline__ float sigf(float x) { return 1.f / (1.f + __expf(-x)); }

// Gate nonlinearity + state update; scatters h into the [nSeq,T,512] layer
// output (fw -> cols 0:256 masked by len, bw -> cols 256:512 un-reversed).
__global__ void k_lstm_gate(const float* __restrict__ g, const float* __restrict__ bias,
                            float* __restrict__ cst, half_t* __restrict__ hst,
                            half_t* __restrict__ out16, float* __restrict__ out32,
                            const int* __restrict__ lens, int nSeq, int T, int t, int dir) {
  long i = (long)blockIdx.x * blockDim.x + threadIdx.x;
  if (i >= (long)nSeq * HH_) return;
  const int r = (int)(i >> 8), j = (int)(i & (HH_ - 1));
  const float* gr = g + (long)r * G4_;
  const float ig = sigf(gr[j] + bias[j]);
  const float fg = sigf(gr[HH_ + j] + bias[HH_ + j]);
  const float gg = tanhf(gr[2 * HH_ + j] + bias[2 * HH_ + j]);
  const float og = sigf(gr[3 * HH_ + j] + bias[3 * HH_ + j]);
  const float c = fg * cst[i] + ig * gg;
  const float h = og * tanhf(c);
  cst[i] = c;
  hst[i] = (half_t)h;
  const int len = lens[r];
  if (dir == 0) {
    const float hw = (t < len) ? h : 0.f;
    long o = ((long)r * T + t) * NH_ + j;
    out16[o] = (half_t)hw;
    if (out32) out32[o] = hw;
  } else if (t < len) {
    long o = ((long)r * T + (len - 1 - t)) * NH_ + HH_ + j;
    out16[o] = (half_t)h;
    if (out32) out32[o] = h;
  }
}

// --------------------------- attention / misc -----------------------------
__global__ void k_mask_att(float* __restrict__ att, const int* __restrict__ qlen,
                           const int* __restrict__ clen, int C, int T) {
  long i = (long)blockIdx.x * blockDim.x + threadIdx.x;
  if (i >= (long)BB * C * T) return;
  const int t = (int)(i % T);
  const int c = (int)((i / T) % C);
  const int b = (int)(i / ((long)T * C));
  const bool ok = (t < qlen[b]) && (!clen || c < clen[b]);
  if (!ok) att[i] = -100.f;
}

__global__ void k_softmax(float* __restrict__ x, int rows, int T) {
  const int row = blockIdx.x * (blockDim.x >> 5) + (threadIdx.x >> 5);
  const int lane = threadIdx.x & 31;
  if (row >= rows) return;
  float* p = x + (long)row * T;
  float mx = -3.4e38f;
  for (int i = lane; i < T; i += 32) mx = fmaxf(mx, p[i]);
  for (int o = 16; o > 0; o >>= 1) mx = fmaxf(mx, __shfl_xor(mx, o, 32));
  float s = 0.f;
  for (int i = lane; i < T; i += 32) { float e = __expf(p[i] - mx); p[i] = e; s += e; }
  for (int o = 16; o > 0; o >>= 1) s += __shfl_xor(s, o, 32);
  const float inv = 1.f / s;
  for (int i = lane; i < T; i += 32) p[i] *= inv;
}

__global__ void k_reduce_c(const float* __restrict__ prob, float* __restrict__ pc) {
  long i = (long)blockIdx.x * blockDim.x + threadIdx.x;
  if (i >= (long)BB * TT) return;
  const int b = (int)(i / TT), t = (int)(i % TT);
  const float* p = prob + (long)b * CM_ * TT + t;
  float s = 0.f;
  for (int c = 0; c < CM_; ++c) s += p[(long)c * TT];
  pc[i] = s;
}

__global__ void k_qwn(const float* __restrict__ pc, const float* __restrict__ qenc,
                      float* __restrict__ o32, half_t* __restrict__ o16) {
  long i = (long)blockIdx.x * blockDim.x + threadIdx.x;
  if (i >= (long)BB * NH_) return;
  const int b = (int)(i >> 9), h = (int)(i & (NH_ - 1));
  const float* pcb = pc + (long)b * TT;
  const float* qb = qenc + (long)b * TT * NH_ + h;
  float s = 0.f;
  for (int t = 0; t < TT; ++t) s += pcb[t] * qb[(long)t * NH_];
  o32[i] = s; o16[i] = (half_t)s;
}

__global__ void k_tanh_add(float* __restrict__ d, const float* __restrict__ a,
                           const float* __restrict__ b, long n) {
  long i = (long)blockIdx.x * blockDim.x + threadIdx.x;
  if (i < n) d[i] = tanhf(a[i] + (b ? b[i] : 0.f));
}

__global__ void k_offsets(const int* __restrict__ cl, int* __restrict__ offs) {
  if (blockIdx.x == 0 && threadIdx.x == 0) {
    int a = 0;
    for (int i = 0; i < BB; ++i) { offs[i] = a; a += cl[i]; }
  }
}

__global__ void k_colenc(const half_t* __restrict__ hid, const int* __restrict__ nlen,
                         const int* __restrict__ offs, const int* __restrict__ clen,
                         int total, float* __restrict__ e32, half_t* __restrict__ e16) {
  long i = (long)blockIdx.x * blockDim.x + threadIdx.x;
  if (i >= (long)BB * CM_ * NH_) return;
  const int h = (int)(i % NH_);
  const int c = (int)((i / NH_) % CM_);
  const int b = (int)(i / ((long)NH_ * CM_));
  float v = 0.f;
  if (c < clen[b]) {
    long n = offs[b] + c; if (n >= total) n = total - 1;
    const int L = nlen[n];
    v = (float)hid[((long)n * LN_ + (L - 1)) * NH_ + h];
  }
  e32[i] = v; e16[i] = (half_t)v;
}

__global__ void k_gather_sel(const float* __restrict__ enc, const int* __restrict__ sel,
                             float* __restrict__ o32, half_t* __restrict__ o16) {
  long i = (long)blockIdx.x * blockDim.x + threadIdx.x;
  if (i >= (long)BB * KP_ * NH_) return;
  const int h = (int)(i % NH_);
  const int k = (int)((i / NH_) % KP_);
  const int b = (int)(i / ((long)NH_ * KP_));
  float v = 0.f;
  if (k < NS_) {
    const int c = sel[b * NS_ + k];
    v = enc[((long)b * CM_ + c) * NH_ + h];
  }
  o32[i] = v; o16[i] = (half_t)v;
}

__global__ void k_head(const float* __restrict__ x, const float* __restrict__ W,
                       const float* __restrict__ bias, float* __restrict__ out,
                       int rows, int dout) {
  long i = (long)blockIdx.x * blockDim.x + threadIdx.x;
  if (i >= (long)rows * dout) return;
  const int r = (int)(i / dout), o = (int)(i % dout);
  const float* xr = x + (long)r * NH_;
  const float* wr = W + (long)o * NH_;
  float s = bias[o];
  for (int d = 0; d < NH_; ++d) s += xr[d] * wr[d];
  out[i] = s;
}

__global__ void k_col_head(const float* __restrict__ x, const float* __restrict__ W,
                           const float* __restrict__ bias, const int* __restrict__ clen,
                           float* __restrict__ out) {
  long i = (long)blockIdx.x * blockDim.x + threadIdx.x;
  if (i >= (long)BB * CM_) return;
  const int b = (int)(i / CM_), c = (int)(i % CM_);
  float s = -100.f;
  if (c < clen[b]) {
    const float* xr = x + i * (long)NH_;
    s = bias[0];
    for (int d = 0; d < NH_; ++d) s += xr[d] * W[d];
  }
  out[i] = s;
}

__global__ void k_head_agg(const float* __restrict__ x, const float* __restrict__ W,
                           const float* __restrict__ bias, float* __restrict__ out,
                           int dout) {
  long i = (long)blockIdx.x * blockDim.x + threadIdx.x;
  if (i >= (long)BB * NS_ * dout) return;
  const int o = (int)(i % dout);
  const int k = (int)((i / dout) % NS_);
  const int b = (int)(i / ((long)dout * NS_));
  const float* xr = x + ((long)b * KP_ + k) * NH_;
  const float* wr = W + (long)o * NH_;
  float s = bias[o];
  for (int d = 0; d < NH_; ++d) s += xr[d] * wr[d];
  out[i] = s;
}

// -------------------------- param leaf indices ----------------------------
// Top-level dict in insertion order; "params" pytree flattened jax-style
// (dict keys sorted, lists in order; lstm layer = {bw:{Whh,Wih,b}, fw:{...}}).
enum {
  P_AGG_ATT_W = 6,      P_AGG_ATT_B,
  P_AGG_NUM_ATT_W,      P_AGG_NUM_ATT_B,     // 8,9
  P_AGG_NUM_OUT_W,      P_AGG_NUM_OUT_B,     // 10,11
  P_AGG_NUM_OUT_C_W,    P_AGG_NUM_OUT_C_B,   // 12,13
  P_AGG_NUM_OUT_Q_W,    P_AGG_NUM_OUT_Q_B,   // 14,15
  P_AGG_OUT_W,          P_AGG_OUT_B,         // 16,17
  P_AGG_OUT_C_W,        P_AGG_OUT_C_B,       // 18,19
  P_AGG_OUT_Q_W,        P_AGG_OUT_Q_B,       // 20,21
  P_COL_LSTM = 22,                           // 12 leaves
  P_COL_NUM_OUT_W = 34, P_COL_NUM_OUT_B,
  P_COL_NUM_OUT_Q_W = 36, P_COL_NUM_OUT_Q_B,
  P_COL_OUT_W = 38,     P_COL_OUT_B,
  P_COL_OUT_C_W = 40,   P_COL_OUT_C_B,
  P_COL_OUT_Q_W = 42,   P_COL_OUT_Q_B,
  P_Q_ATT_W = 44,       P_Q_ATT_B,
  P_Q_LSTM = 46,                             // 12 leaves
  P_Q_NUM_ATT_W = 58,   P_Q_NUM_ATT_B
};

extern "C" void kernel_launch(void* const* d_in, const int* in_sizes, int n_in,
                              void* d_out, int out_size, void* d_ws, size_t ws_size,
                              hipStream_t stream) {
  (void)n_in; (void)out_size; (void)ws_size;
  const float* q_emb   = (const float*)d_in[0];
  const int*   q_len   = (const int*)  d_in[1];
  const float* col_emb = (const float*)d_in[2];
  const int*   col_len = (const int*)  d_in[3];
  const int*   nm_len  = (const int*)  d_in[4];
  const int*   gt_sel  = (const int*)  d_in[5];
  float*       d_o     = (float*)d_out;
  #define PF(i) ((const float*)d_in[(i)])

  const int total = in_sizes[4];            // number of column names
  const int TOTP  = (total + 15) & ~15;     // padded to WMMA M granularity

  char* wsb = (char*)d_ws; size_t off = 0;
  auto alloc = [&](size_t bytes) -> void* {
    void* p = wsb + off;
    off = (off + bytes + 255) & ~(size_t)255;
    return p;
  };
  #define AL(T, n) (T*)alloc((size_t)(n) * sizeof(T))

  // LSTM packed weights (layer0: K=32+256, layer1: K=512+256)
  half_t* wq[4]; half_t* wc[4];
  for (int l = 0; l < 2; ++l) {
    const long dK = (l ? NH_ : 32) + HH_;
    for (int d = 0; d < 2; ++d) { wq[l*2+d] = AL(half_t, (long)G4_*dK);
                                  wc[l*2+d] = AL(half_t, (long)G4_*dK); }
  }
  const long S512 = (long)NH_ * NH_;
  half_t *w_qnum = AL(half_t,S512), *w_qatt = AL(half_t,S512),
         *w_anatt= AL(half_t,S512), *w_aatt = AL(half_t,S512),
         *w_cnoq = AL(half_t,S512), *w_coq  = AL(half_t,S512),
         *w_coc  = AL(half_t,S512), *w_anoq = AL(half_t,S512),
         *w_anoc = AL(half_t,S512), *w_aoq  = AL(half_t,S512),
         *w_aoc  = AL(half_t,S512);

  half_t* abuf = AL(half_t, (long)TOTP * (NH_ + HH_));
  float*  gtmp = AL(float,  (long)TOTP * G4_);
  float*  cst  = AL(float,  (long)TOTP * HH_);
  half_t* hst  = AL(half_t, (long)TOTP * HH_);
  half_t* qxa  = AL(half_t, (long)BB * TT * NH_);
  half_t* qxb  = AL(half_t, (long)BB * TT * NH_);   // f16 q_enc
  float*  qenc = AL(float,  (long)BB * TT * NH_);
  half_t* cxa  = AL(half_t, (long)TOTP * LN_ * NH_);
  half_t* cxb  = AL(half_t, (long)TOTP * LN_ * NH_);
  float*  ce32v= AL(float,  (long)BB * CM_ * NH_);  // col_enc f32
  half_t* ce16v= AL(half_t, (long)BB * CM_ * NH_);
  float*  Qb   = AL(float,  (long)BB * TT * NH_);
  half_t* Qb16 = AL(half_t, (long)BB * TT * NH_);
  float*  att  = AL(float,  (long)BB * CM_ * TT);
  float*  pc   = AL(float,  (long)BB * TT);
  float*  qwn  = AL(float,  (long)BB * NH_);
  half_t* qwn16= AL(half_t, (long)BB * NH_);
  float*  qW   = AL(float,  (long)BB * CM_ * NH_);
  half_t* qW16 = AL(half_t, (long)BB * CM_ * NH_);
  float*  t2   = AL(float,  (long)BB * CM_ * NH_);
  float*  t3   = AL(float,  (long)BB * CM_ * NH_);
  float*  sel32= AL(float,  (long)BB * KP_ * NH_);  // gathered col_emb
  half_t* sel16= AL(half_t, (long)BB * KP_ * NH_);
  int*    offs = AL(int, BB);

  auto g1 = [](long n) { return dim3((unsigned)((n + 255) / 256)); };
  auto zero = [&](void* p, long bytes) {
    long n = bytes >> 2;
    k_zero<<<g1(n), 256, 0, stream>>>((uint32_t*)p, n);
  };
  auto cvt = [&](const float* s, half_t* d, long n) {
    k_cvt<<<g1(n), 256, 0, stream>>>(s, d, n);
  };
  auto gemm_tn = [&](const half_t* A, const half_t* W, const float* bias,
                     float* C, half_t* C16, int M, int N, int K,
                     long sA, long sW, long sC, int batch) {
    if (N % 256 == 0) {        // 16x64 strip per wave (LSTM gates, projections)
      dim3 g((unsigned)(M / 16), (unsigned)(N / 256), (unsigned)batch);
      k_gemm_tn<4><<<g, 128, 0, stream>>>(A, W, bias, C, C16, M, N, K, sA, sW, sC);
    } else {                   // attention scores (N = 128)
      dim3 g((unsigned)(M / 16), (unsigned)((N + 63) / 64), (unsigned)batch);
      k_gemm_tn<1><<<g, 128, 0, stream>>>(A, W, bias, C, C16, M, N, K, sA, sW, sC);
    }
  };
  auto gemm_nn = [&](const float* A, const half_t* B, float* C, half_t* C16,
                     int M, int N, int K, long sA, long sB, long sC, int batch) {
    if (N % 256 == 0) {
      dim3 g((unsigned)(M / 16), (unsigned)(N / 256), (unsigned)batch);
      k_gemm_nn<4><<<g, 128, 0, stream>>>(A, B, C, C16, M, N, K, sA, sB, sC);
    } else {
      dim3 g((unsigned)(M / 16), (unsigned)((N + 63) / 64), (unsigned)batch);
      k_gemm_nn<1><<<g, 128, 0, stream>>>(A, B, C, C16, M, N, K, sA, sB, sC);
    }
  };
  auto lstm_dir = [&](const float* s32, const half_t* s16, int nSeq, int Mpad, int T,
                      int dinSrc, int din_p, const int* lens, const half_t* wcat,
                      const float* bias, half_t* out16, float* out32, int dir) {
    const int dK = din_p + HH_;
    zero(cst, (long)Mpad * HH_ * 4);
    zero(hst, (long)Mpad * HH_ * 2);
    for (int t = 0; t < T; ++t) {
      const long np = (long)Mpad * dK;
      k_pack_step<<<g1(np), 256, 0, stream>>>(s32, s16, hst, abuf, lens,
                                              nSeq, Mpad, T, dinSrc, din_p, t, dir);
      gemm_tn(abuf, wcat, nullptr, gtmp, nullptr, Mpad, G4_, dK, 0, 0, 0, 1);
      const long ng = (long)nSeq * HH_;
      k_lstm_gate<<<g1(ng), 256, 0, stream>>>(gtmp, bias, cst, hst, out16, out32,
                                              lens, nSeq, T, t, dir);
    }
  };
  auto bilstm = [&](int base, const float* emb, const int* lens, int nSeq, int Mpad,
                    int T, half_t* const* wcat4, half_t* xa, half_t* xb, float* e32) {
    for (int l = 0; l < 2; ++l) {
      const int din = l ? NH_ : NW_, din_p = l ? NH_ : 32;
      for (int d = 0; d < 2; ++d) {                 // d: 0=fw(+3), 1=bw(+0)
        const int lb = base + l * 6 + (d == 0 ? 3 : 0);
        const long n = (long)G4_ * (din_p + HH_);
        k_pack_wcat<<<g1(n), 256, 0, stream>>>(PF(lb + 1), PF(lb + 0),
                                               wcat4[l * 2 + d], din, din_p);
      }
    }
    lstm_dir(emb, nullptr, nSeq, Mpad, T, NW_, 32, lens, wcat4[0], PF(base + 5), xa, nullptr, 0);
    lstm_dir(emb, nullptr, nSeq, Mpad, T, NW_, 32, lens, wcat4[1], PF(base + 2), xa, nullptr, 1);
    lstm_dir(nullptr, xa, nSeq, Mpad, T, NH_, NH_, lens, wcat4[2], PF(base + 11), xb, e32, 0);
    lstm_dir(nullptr, xa, nSeq, Mpad, T, NH_, NH_, lens, wcat4[3], PF(base + 8),  xb, e32, 1);
  };

  // ---- weight conversion + scratch init (ws is poisoned every timing run) --
  cvt(PF(P_Q_NUM_ATT_W), w_qnum, S512);  cvt(PF(P_Q_ATT_W), w_qatt, S512);
  cvt(PF(P_AGG_NUM_ATT_W), w_anatt, S512); cvt(PF(P_AGG_ATT_W), w_aatt, S512);
  cvt(PF(P_COL_NUM_OUT_Q_W), w_cnoq, S512);
  cvt(PF(P_COL_OUT_Q_W), w_coq, S512);   cvt(PF(P_COL_OUT_C_W), w_coc, S512);
  cvt(PF(P_AGG_NUM_OUT_Q_W), w_anoq, S512); cvt(PF(P_AGG_NUM_OUT_C_W), w_anoc, S512);
  cvt(PF(P_AGG_OUT_Q_W), w_aoq, S512);   cvt(PF(P_AGG_OUT_C_W), w_aoc, S512);
  zero(qxa, (long)BB * TT * NH_ * 2);  zero(qxb, (long)BB * TT * NH_ * 2);
  zero(qenc, (long)BB * TT * NH_ * 4);
  zero(cxa, (long)TOTP * LN_ * NH_ * 2); zero(cxb, (long)TOTP * LN_ * NH_ * 2);

  // ---- bi-LSTM encoders --------------------------------------------------
  bilstm(P_Q_LSTM,   q_emb,   q_len,  BB,    BB,   TT,  wq, qxa, qxb, qenc);
  bilstm(P_COL_LSTM, col_emb, nm_len, total, TOTP, LN_, wc, cxa, cxb, nullptr);

  // ---- col_enc: last valid hidden per name, scattered to [B,48,512] -------
  k_offsets<<<1, 1, 0, stream>>>(col_len, offs);
  k_colenc<<<g1((long)BB * CM_ * NH_), 256, 0, stream>>>(cxb, nm_len, offs, col_len,
                                                         total, ce32v, ce16v);

  // ---- predict col num ----------------------------------------------------
  gemm_tn(qxb, w_qnum, PF(P_Q_NUM_ATT_B), Qb, Qb16, BB * TT, NH_, NH_, 0, 0, 0, 1);
  gemm_tn(ce16v, Qb16, nullptr, att, nullptr, CM_, TT, NH_,
          (long)CM_ * NH_, (long)TT * NH_, (long)CM_ * TT, BB);
  k_mask_att<<<g1((long)BB * CM_ * TT), 256, 0, stream>>>(att, q_len, col_len, CM_, TT);
  k_softmax<<<(BB * CM_ + 7) / 8, 256, 0, stream>>>(att, BB * CM_, TT);
  k_reduce_c<<<g1((long)BB * TT), 256, 0, stream>>>(att, pc);
  k_qwn<<<g1((long)BB * NH_), 256, 0, stream>>>(pc, qenc, qwn, qwn16);
  gemm_tn(qwn16, w_cnoq, PF(P_COL_NUM_OUT_Q_B), t2, nullptr, BB, NH_, NH_, 0, 0, 0, 1);
  k_tanh_add<<<g1((long)BB * NH_), 256, 0, stream>>>(t2, t2, nullptr, (long)BB * NH_);
  k_head<<<g1((long)BB * 5), 256, 0, stream>>>(t2, PF(P_COL_NUM_OUT_W),
                                               PF(P_COL_NUM_OUT_B), d_o, BB, 5);

  // ---- predict columns ----------------------------------------------------
  gemm_tn(qxb, w_qatt, PF(P_Q_ATT_B), Qb, Qb16, BB * TT, NH_, NH_, 0, 0, 0, 1);
  gemm_tn(ce16v, Qb16, nullptr, att, nullptr, CM_, TT, NH_,
          (long)CM_ * NH_, (long)TT * NH_, (long)CM_ * TT, BB);
  k_mask_att<<<g1((long)BB * CM_ * TT), 256, 0, stream>>>(att, q_len, nullptr, CM_, TT);
  k_softmax<<<(BB * CM_ + 7) / 8, 256, 0, stream>>>(att, BB * CM_, TT);
  gemm_nn(att, qxb, qW, qW16, CM_, NH_, TT,
          (long)CM_ * TT, (long)TT * NH_, (long)CM_ * NH_, BB);
  gemm_tn(qW16, w_coq, PF(P_COL_OUT_Q_B), t2, nullptr, BB * CM_, NH_, NH_, 0, 0, 0, 1);
  gemm_tn(ce16v, w_coc, PF(P_COL_OUT_C_B), t3, nullptr, BB * CM_, NH_, NH_, 0, 0, 0, 1);
  k_tanh_add<<<g1((long)BB * CM_ * NH_), 256, 0, stream>>>(t2, t2, t3, (long)BB * CM_ * NH_);
  k_col_head<<<g1((long)BB * CM_), 256, 0, stream>>>(t2, PF(P_COL_OUT_W),
                                                     PF(P_COL_OUT_B), col_len, d_o + 640);

  // ---- agg blocks ----------------------------------------------------------
  k_gather_sel<<<g1((long)BB * KP_ * NH_), 256, 0, stream>>>(ce32v, gt_sel, sel32, sel16);
  struct Blk { half_t *hAtt, *hOq, *hOc; int bAtt, bOq, bOc, wOut, bOut, dout; long ooff; };
  const Blk blks[2] = {
    { w_anatt, w_anoq, w_anoc, P_AGG_NUM_ATT_B, P_AGG_NUM_OUT_Q_B, P_AGG_NUM_OUT_C_B,
      P_AGG_NUM_OUT_W, P_AGG_NUM_OUT_B, 5, 640 + 6144 },
    { w_aatt,  w_aoq,  w_aoc,  P_AGG_ATT_B,     P_AGG_OUT_Q_B,     P_AGG_OUT_C_B,
      P_AGG_OUT_W,     P_AGG_OUT_B,     6, 640 + 6144 + 3200 }
  };
  for (int bi = 0; bi < 2; ++bi) {
    const Blk& bk = blks[bi];
    gemm_tn(qxb, bk.hAtt, PF(bk.bAtt), Qb, Qb16, BB * TT, NH_, NH_, 0, 0, 0, 1);
    gemm_tn(sel16, Qb16, nullptr, att, nullptr, KP_, TT, NH_,
            (long)KP_ * NH_, (long)TT * NH_, (long)KP_ * TT, BB);
    k_mask_att<<<g1((long)BB * KP_ * TT), 256, 0, stream>>>(att, q_len, nullptr, KP_, TT);
    k_softmax<<<(BB * KP_ + 7) / 8, 256, 0, stream>>>(att, BB * KP_, TT);
    gemm_nn(att, qxb, qW, qW16, KP_, NH_, TT,
            (long)KP_ * TT, (long)TT * NH_, (long)KP_ * NH_, BB);
    gemm_tn(qW16, bk.hOq, PF(bk.bOq), t2, nullptr, BB * KP_, NH_, NH_, 0, 0, 0, 1);
    gemm_tn(sel16, bk.hOc, PF(bk.bOc), t3, nullptr, BB * KP_, NH_, NH_, 0, 0, 0, 1);
    k_tanh_add<<<g1((long)BB * KP_ * NH_), 256, 0, stream>>>(t2, t2, t3,
                                                             (long)BB * KP_ * NH_);
    k_head_agg<<<g1((long)BB * NS_ * bk.dout), 256, 0, stream>>>(t2, PF(bk.wOut),
                                                                 PF(bk.bOut),
                                                                 d_o + bk.ooff, bk.dout);
  }
  #undef AL
  #undef PF
}
